// SMN_69191923138674
// MI455X (gfx1250) — compile-verified, matrix operand order checked by reading, not run on hardware
//
#include <hip/hip_runtime.h>
#include <hip/hip_bf16.h>
#include <math.h>

// ---------------- dims ----------------
#define S_  48
#define M_  128
#define DIN_ 128
#define DM_ 128
#define H_  4
#define DH_ 128
#define HD_ 512           // H*DH
#define L_  2
#define ALPHA_ 0.3f
#define INV_SQRT_DH_ 0.08838834764831843f  // 1/sqrt(128)

// GEMM tiling
#define BM_ 64
#define BN_ 64
#define BK_ 32
#define LDSLD_ 40   // f16 row stride (80B): padded, 16B-aligned

typedef __attribute__((ext_vector_type(16))) _Float16 v16h;
typedef __attribute__((ext_vector_type(8)))  float    v8f;

// ---------------- WMMA helpers (gfx1250, wave32) ----------------
// Fragment with K contiguous in memory: element (i, k) at src[i*ld + k],
// i = lane&15 (row for A, col for B^T), K halves split across lane halves.
__device__ __forceinline__ v16h frag_contig_h(const _Float16* src, int ld, int lane) {
  int i  = lane & 15;
  int kh = (lane >> 4) & 1;
  const _Float16* p = src + i * ld + kh * 8;
  v16h r;
#pragma unroll
  for (int e = 0; e < 16; ++e) {
    int k = (e < 8 ? 0 : 16) + (e & 7);
    r[e] = p[k];
  }
  return r;
}

__device__ __forceinline__ v8f wmma32(v16h a, v16h b, v8f c) {
  return __builtin_amdgcn_wmma_f32_16x16x32_f16(false, a, false, b, (short)0, c, false, false);
}

__device__ __forceinline__ float gelu_f(float x) {
  return 0.5f * x * (1.0f + erff(x * 0.7071067811865476f));
}

// ---------------- kernel: item mask + set sizes ----------------
__global__ void SMN_mask_size_k(const float* __restrict__ x, const int* __restrict__ xs,
                                float* __restrict__ maskb, float* __restrict__ sizef) {
  int idx = blockIdx.x * blockDim.x + threadIdx.x;
  if (idx < S_ * M_) {
    const float* p = x + (long long)idx * DIN_;
    int nz = 0;
    for (int d = 0; d < DIN_; ++d) nz += (p[d] != 0.0f);
    maskb[idx] = (nz > 0) ? 1.0f : 0.0f;
  }
  if (idx < S_) {
    int v = xs[idx];
    sizef[idx] = (float)(v < 1 ? 1 : v);
  }
}

// ---------------- LDS-tiled batched WMMA GEMM ----------------
// Block tile 64x64, K-chunk 32, 8 waves; each wave: one A-frag, two B-frags,
// two v_wmma per chunk. B panel staged transposed (K-contiguous) so both the
// normal and B^T cases use the same vectorized LDS fragment loader.
// epi: 0 none | 1 gelu*mask | 2 *mask | 3 resid+gelu*mask | 4 resid+*mask
__global__ __launch_bounds__(256)
void SMN_wmma_gemm_k(const float* __restrict__ A, int lda, long long sAy, long long sAz,
                     const float* __restrict__ B, int ldb, long long sBy, long long sBz, int bT,
                     float* __restrict__ C, int ldc, long long sCy, long long sCz,
                     const float* __restrict__ resid, const float* __restrict__ maskrow,
                     int Mr, int N, int K, int epi, float scale) {
  __shared__ _Float16 As[BM_ * LDSLD_];
  __shared__ _Float16 Bs[BN_ * LDSLD_];
  A += blockIdx.y * sAy + blockIdx.z * sAz;
  B += blockIdx.y * sBy + blockIdx.z * sBz;
  C += blockIdx.y * sCy + blockIdx.z * sCz;
  const int nBN = N / BN_;
  const int row0 = (blockIdx.x / nBN) * BM_;
  const int col0 = (blockIdx.x % nBN) * BN_;
  const int tid = threadIdx.x, wid = tid >> 5, lane = tid & 31;
  const int tm  = wid >> 1;          // 0..3 : 16-row tile within block
  const int tn0 = (wid & 1) * 2;     // 0 or 2: first of two 16-col tiles
  v8f acc0 = {}, acc1 = {};

  for (int k0 = 0; k0 < K; k0 += BK_) {
    // ---- stage A panel: BM_ x BK_ (row-major, K contiguous) ----
    {
      int idx = tid * 8;                      // 2048 elems / 256 threads
      int r = idx >> 5, kk = idx & 31;        // kk in {0,8,16,24}
      const float* g = A + (long long)(row0 + r) * lda + k0 + kk;
      _Float16* dst = As + r * LDSLD_ + kk;
#pragma unroll
      for (int j = 0; j < 8; ++j) dst[j] = (_Float16)g[j];
      if (k0 + BK_ < K) __builtin_prefetch(g + BK_, 0, 1);   // global_prefetch_b8
    }
    // ---- stage B panel as Bs[n][k] (K contiguous in LDS) ----
    if (bT) {                                  // global: element (k,n) at B[n*ldb+k]
      int idx = tid * 8;
      int n = idx >> 5, kk = idx & 31;
      const float* g = B + (long long)(col0 + n) * ldb + k0 + kk;
      _Float16* dst = Bs + n * LDSLD_ + kk;
#pragma unroll
      for (int j = 0; j < 8; ++j) dst[j] = (_Float16)g[j];
    } else {                                   // global: element (k,n) at B[k*ldb+n]
      int idx = tid * 8;
      int kk = idx >> 6, n = idx & 63;         // n in {0,8,...,56}
      const float* g = B + (long long)(k0 + kk) * ldb + col0 + n;
#pragma unroll
      for (int j = 0; j < 8; ++j) Bs[(n + j) * LDSLD_ + kk] = (_Float16)g[j];
    }
    __syncthreads();
    v16h af = frag_contig_h(As + (tm * 16) * LDSLD_, LDSLD_, lane);
    v16h b0 = frag_contig_h(Bs + (tn0 * 16) * LDSLD_, LDSLD_, lane);
    v16h b1 = frag_contig_h(Bs + ((tn0 + 1) * 16) * LDSLD_, LDSLD_, lane);
    acc0 = wmma32(af, b0, acc0);
    acc1 = wmma32(af, b1, acc1);
    __syncthreads();
  }

  // ---- epilogue: two 16x16 tiles per wave ----
  const int col = lane & 15;
  const int r0  = ((lane >> 4) ? 8 : 0) + row0 + tm * 16;
#pragma unroll
  for (int tt = 0; tt < 2; ++tt) {
    v8f acc = tt ? acc1 : acc0;
    int gc = col0 + (tn0 + tt) * 16 + col;
#pragma unroll
    for (int j = 0; j < 8; ++j) {
      int gr = r0 + j;
      float v = acc[j] * scale;
      float o;
      switch (epi) {
        case 1: o = gelu_f(v) * maskrow[gr]; break;
        case 2: o = v * maskrow[gr]; break;
        case 3: o = resid[(long long)gr * ldc + gc] + gelu_f(v) * maskrow[gr]; break;
        case 4: o = resid[(long long)gr * ldc + gc] + v * maskrow[gr]; break;
        default: o = v; break;
      }
      C[(long long)gr * ldc + gc] = o;
    }
  }
}

// ---------------- masked softmax (one wave32 per row of 128) ----------------
__global__ void SMN_softmax_k(float* __restrict__ s) {
  long long row = blockIdx.x;
  float* p = s + row * 128;
  int lane = threadIdx.x;
  float v0 = p[lane], v1 = p[lane + 32], v2 = p[lane + 64], v3 = p[lane + 96];
  float mx = fmaxf(fmaxf(v0, v1), fmaxf(v2, v3));
#pragma unroll
  for (int off = 16; off > 0; off >>= 1) mx = fmaxf(mx, __shfl_xor(mx, off, 32));
  float e0 = (v0 != 0.0f) ? expf(v0 - mx) : 0.0f;
  float e1 = (v1 != 0.0f) ? expf(v1 - mx) : 0.0f;
  float e2 = (v2 != 0.0f) ? expf(v2 - mx) : 0.0f;
  float e3 = (v3 != 0.0f) ? expf(v3 - mx) : 0.0f;
  float se = e0 + e1 + e2 + e3;
#pragma unroll
  for (int off = 16; off > 0; off >>= 1) se += __shfl_xor(se, off, 32);
  float inv = 1.0f / (se + 1e-10f);
  p[lane] = e0 * inv; p[lane + 32] = e1 * inv; p[lane + 64] = e2 * inv; p[lane + 96] = e3 * inv;
}

// ---------------- set_norm: one 128-thread block per set ----------------
__global__ void SMN_setnorm_k(const float* __restrict__ t, float* __restrict__ out,
                              const float* __restrict__ sizef) {
  __shared__ float sh[M_];
  int s = blockIdx.x, m = threadIdx.x;
  const float* base = t + (long long)s * M_ * DM_;
  float rsum = 0.0f;
  for (int d = 0; d < DM_; ++d) rsum += base[m * DM_ + d];
  float rowmask = (rsum != 0.0f) ? 1.0f : 0.0f;
  sh[m] = rsum; __syncthreads();
  for (int off = 64; off > 0; off >>= 1) { if (m < off) sh[m] += sh[m + off]; __syncthreads(); }
  float denom = sizef[s] * (float)DM_;
  float mean = sh[0] / denom;
  __syncthreads();
  float vsum = 0.0f;
  for (int d = 0; d < DM_; ++d) { float dv = base[m * DM_ + d] - mean; vsum += dv * dv; }
  sh[m] = vsum * rowmask; __syncthreads();
  for (int off = 64; off > 0; off >>= 1) { if (m < off) sh[m] += sh[m + off]; __syncthreads(); }
  float var = sh[0] / denom;
  float inv = rowmask / (sqrtf(var) + 1e-8f);
  float* op = out + (long long)s * M_ * DM_;
  for (int d = 0; d < DM_; ++d) op[m * DM_ + d] = (base[m * DM_ + d] - mean) * inv;
}

// ---------------- cross-set Gram + leaky + block sum (dominant GEMM) -------
// grid = (H, Sx, Sy), 256 threads (8 waves); 64KB LDS stages both Z tiles (f16)
__global__ __launch_bounds__(256)
void SMN_cross_k(const float* __restrict__ z, float* __restrict__ gsum) {
  __shared__ char smem[65536];
  _Float16* Ay = (_Float16*)smem;                 // [128][128] f16
  _Float16* Bx = Ay + 128 * 128;                  // [128][128] f16
  int h = blockIdx.x, sx = blockIdx.y, sy = blockIdx.z;
  const float* zy = z + (long long)sy * M_ * HD_ + h * DH_;
  const float* zx = z + (long long)sx * M_ * HD_ + h * DH_;
  for (int i = threadIdx.x; i < M_ * DH_; i += 256) {
    int m = i >> 7, d = i & 127;
    Ay[m * 128 + d] = (_Float16)zy[(long long)m * HD_ + d];
    Bx[m * 128 + d] = (_Float16)zx[(long long)m * HD_ + d];
  }
  __syncthreads();
  int wid = threadIdx.x >> 5, lane = threadIdx.x & 31;
  float part = 0.0f;
  for (int t = wid; t < 64; t += 8) {     // 8x8 tiles of 16x16
    int tm = t >> 3, tn = t & 7;
    v8f acc = {};
#pragma unroll
    for (int k0 = 0; k0 < DH_; k0 += 32) {
      v16h a = frag_contig_h(Ay + tm * 16 * 128 + k0, 128, lane);
      v16h b = frag_contig_h(Bx + tn * 16 * 128 + k0, 128, lane);  // B^T path
      acc = wmma32(a, b, acc);
    }
#pragma unroll
    for (int j = 0; j < 8; ++j) {
      float v = acc[j] * INV_SQRT_DH_;
      part += (v >= 0.0f) ? v : ALPHA_ * v;   // leaky relu then sum
    }
  }
  __syncthreads();                 // done reading Ay/Bx; reuse LDS for reduce
  float* red = (float*)smem;
  red[threadIdx.x] = part; __syncthreads();
  for (int off = 128; off > 0; off >>= 1) {
    if (threadIdx.x < off) red[threadIdx.x] += red[threadIdx.x + off];
    __syncthreads();
  }
  if (threadIdx.x == 0) gsum[((long long)sy * S_ + sx) * H_ + h] = red[0];
}

// ---------------- finalize: /size[i] /M then @ w2 ----------------
__global__ void SMN_final_k(const float* __restrict__ gsum, const float* __restrict__ sizef,
                            const float* __restrict__ w2, float* __restrict__ out) {
  int idx = blockIdx.x * blockDim.x + threadIdx.x;
  if (idx >= S_ * S_) return;
  int i = idx % S_;                       // sx (size index)
  float inv = 1.0f / (sizef[i] * (float)M_);
  float acc = 0.0f;
#pragma unroll
  for (int h = 0; h < H_; ++h) acc += gsum[(long long)idx * H_ + h] * inv * w2[h];
  out[idx] = acc;
}

// ---------------- host-side launch helpers ----------------
static inline void launch_gemm(hipStream_t stream,
    const float* A, int lda, long long sAy, long long sAz,
    const float* B, int ldb, long long sBy, long long sBz, int bT,
    float* C, int ldc, long long sCy, long long sCz,
    const float* resid, const float* maskrow,
    int Mr, int N, int K, int epi, float scale, int bY, int bZ) {
  dim3 grid((Mr / BM_) * (N / BN_), bY, bZ);
  hipLaunchKernelGGL(SMN_wmma_gemm_k, grid, dim3(256), 0, stream,
      A, lda, sAy, sAz, B, ldb, sBy, sBz, bT, C, ldc, sCy, sCz,
      resid, maskrow, Mr, N, K, epi, scale);
}

extern "C" void kernel_launch(void* const* d_in, const int* in_sizes, int n_in,
                              void* d_out, int out_size, void* d_ws, size_t ws_size,
                              hipStream_t stream) {
  const float* x   = (const float*)d_in[0];
  const int*   xs  = (const int*)d_in[1];
  const float* Wp  = (const float*)d_in[2];
  const float* Wq  = (const float*)d_in[3];   // [L,128,512]
  const float* Wk  = (const float*)d_in[4];
  const float* Wv  = (const float*)d_in[5];
  const float* Wh  = (const float*)d_in[6];   // [L,512,128]
  const float* Wfc = (const float*)d_in[7];   // [L,128,128]
  const float* Wc  = (const float*)d_in[8];   // [128,512]
  const float* w2  = (const float*)d_in[9];   // [4]
  float* out = (float*)d_out;

  float* ws    = (float*)d_ws;
  float* maskb = ws;                     // 6144
  float* sizef = maskb + S_ * M_;        // 48 (pad 64)
  float* gsum  = sizef + 64;             // 9216
  float* h     = gsum + S_ * S_ * H_;    // 786432
  float* t     = h + S_ * M_ * DM_;      // 786432
  float* q     = t + S_ * M_ * DM_;      // 3145728 (also holds o after attn)
  float* kk    = q + S_ * M_ * HD_;      // 3145728
  float* v     = kk + S_ * M_ * HD_;     // 3145728
  float* sb    = v + S_ * M_ * HD_;      // 3145728 (scores; later z)

  const int R = S_ * M_;                 // 6144 rows
  const long long setQ = (long long)M_ * HD_;      // 65536: per-set stride of q/k/v/o
  const long long setS = (long long)H_ * M_ * M_;  // 65536: per-set stride of scores

  // mask + sizes
  hipLaunchKernelGGL(SMN_mask_size_k, dim3((R + 255) / 256), dim3(256), 0, stream,
                     x, xs, maskb, sizef);
  // h = gelu(x @ Wp) * mask
  launch_gemm(stream, x, DIN_, 0, 0, Wp, DM_, 0, 0, 0, h, DM_, 0, 0,
              nullptr, maskb, R, DM_, DIN_, 1, 1.0f, 1, 1);

  for (int l = 0; l < L_; ++l) {
    const float* wql = Wq + (long long)l * DM_ * HD_;
    const float* wkl = Wk + (long long)l * DM_ * HD_;
    const float* wvl = Wv + (long long)l * DM_ * HD_;
    const float* whl = Wh + (long long)l * HD_ * DM_;
    const float* wfl = Wfc + (long long)l * DM_ * DM_;
    // q,k,v = h @ W*  : [6144,128] @ [128,512]
    launch_gemm(stream, h, DM_, 0, 0, wql, HD_, 0, 0, 0, q,  HD_, 0, 0, nullptr, nullptr, R, HD_, DM_, 0, 1.0f, 1, 1);
    launch_gemm(stream, h, DM_, 0, 0, wkl, HD_, 0, 0, 0, kk, HD_, 0, 0, nullptr, nullptr, R, HD_, DM_, 0, 1.0f, 1, 1);
    launch_gemm(stream, h, DM_, 0, 0, wvl, HD_, 0, 0, 0, v,  HD_, 0, 0, nullptr, nullptr, R, HD_, DM_, 0, 1.0f, 1, 1);
    // scores = q @ k^T / sqrt(DH) per (set, head): batched (48, 4)
    launch_gemm(stream, q, HD_, setQ, DH_, kk, HD_, setQ, DH_, 1,
                sb, M_, setS, (long long)M_ * M_, nullptr, nullptr,
                M_, M_, DH_, 0, INV_SQRT_DH_, S_, H_);
    // masked softmax over rows
    hipLaunchKernelGGL(SMN_softmax_k, dim3(S_ * H_ * M_), dim3(32), 0, stream, sb);
    // o = a @ v  (write into q buffer, per-head layout [S,M,H*DH])
    launch_gemm(stream, sb, M_, setS, (long long)M_ * M_, v, HD_, setQ, DH_, 0,
                q, HD_, setQ, DH_, nullptr, nullptr, M_, DH_, M_, 0, 1.0f, S_, H_);
    // t = h + (o @ Wh) * mask
    launch_gemm(stream, q, HD_, 0, 0, whl, DM_, 0, 0, 0, t, DM_, 0, 0,
                h, maskb, R, DM_, HD_, 4, 1.0f, 1, 1);
    hipLaunchKernelGGL(SMN_setnorm_k, dim3(S_), dim3(M_), 0, stream, t, h, sizef);
    // t = h + gelu(h @ Wfc) * mask
    launch_gemm(stream, h, DM_, 0, 0, wfl, DM_, 0, 0, 0, t, DM_, 0, 0,
                h, maskb, R, DM_, DM_, 3, 1.0f, 1, 1);
    hipLaunchKernelGGL(SMN_setnorm_k, dim3(S_), dim3(M_), 0, stream, t, h, sizef);
  }

  // z = h @ Wc : [6144,128] @ [128,512] -> reuse sb
  launch_gemm(stream, h, DM_, 0, 0, Wc, HD_, 0, 0, 0, sb, HD_, 0, 0,
              nullptr, nullptr, R, HD_, DM_, 0, 1.0f, 1, 1);
  // cross-set scores (dominant compute)
  hipLaunchKernelGGL(SMN_cross_k, dim3(H_, S_, S_), dim3(256), 0, stream, sb, gsum);
  // finalize: scale + w2
  hipLaunchKernelGGL(SMN_final_k, dim3((S_ * S_ + 255) / 256), dim3(256), 0, stream,
                     gsum, sizef, w2, out);
}
